// MultiHeadAttention_55722905698530
// MI455X (gfx1250) — matrix operand order repaired and checked by reading, not compiled, not run on hardware
//
#include <hip/hip_runtime.h>

// MI455X / gfx1250 flash-attention + out_proj.
// Memory-bound on attn_bias (128 MB streamed once -> NT loads); matmuls via
// v_wmma_f32_16x16x32_f16 (wave32), f32 accumulation, online softmax.
// K/V chunks staged to LDS with GLOBAL_LOAD_ASYNC_TO_LDS (ASYNCcnt) when the
// toolchain exposes the builtins; padded LDS strides kill bank conflicts.

#define DM   1024   // d_model
#define NH   16
#define HD   64     // head dim
#define TSEQ 1024
#define SSEQ 1024
#define BM   128    // rows per workgroup (8 waves x 16)
#define BN   64     // S-chunk per iteration

#define SKP  68     // sKf row stride (floats): mult-of-4 (16B DS) + bank spread
#define SVP  66     // sVf row stride (floats): mult-of-2 (8B DS) + bank spread
#define SPP  72     // sP  row stride (halves): bank spread, 16B aligned runs

typedef __attribute__((ext_vector_type(16))) _Float16 v16h;
typedef __attribute__((ext_vector_type(8)))  _Float16 v8h;
typedef __attribute__((ext_vector_type(8)))  float    v8f;

#if __has_builtin(__builtin_amdgcn_global_load_async_to_lds_b128) && \
    __has_builtin(__builtin_amdgcn_global_load_async_to_lds_b64)
#define HAVE_ASYNC_LDS 1
#define AS1 __attribute__((address_space(1)))
#define AS3 __attribute__((address_space(3)))
// exact param pointee types per hipcc diagnostic: GCC-style int vectors
typedef __attribute__((__vector_size__(16))) int b128v;
typedef __attribute__((__vector_size__(8)))  int b64v;
__device__ __forceinline__ void wait_async0() {
#if __has_builtin(__builtin_amdgcn_s_wait_asynccnt)
  __builtin_amdgcn_s_wait_asynccnt(0);
#else
  asm volatile("s_wait_asynccnt 0" ::: "memory");
#endif
}
#endif

__device__ __forceinline__ v8f wmma16(v16h a, v16h b, v8f c) {
  return __builtin_amdgcn_wmma_f32_16x16x32_f16(
      false, a, false, b, (short)0, c, false, false);
}

// A-matrix (16x32 f16) per-lane K index for element e (ISA 7.12.2):
// lanes 0-15: e<8 -> K=e, e>=8 -> K=8+e ; lanes 16-31: +8.
__device__ __forceinline__ int a_kmap(int hi, int e) {
  int v = e >> 1, hf = e & 1;
  int base = (v < 4) ? (2 * v + hf) : (16 + 2 * (v - 4) + hf);
  return base + 8 * hi;
}

// ---------------------------------------------------------------------------
// Kernel 1: fused attention with bias/clip/mask + online softmax.
// Writes f16 context [B*T, DM] into workspace.
// grid = B*H*(T/BM) = 256 blocks of 256 threads (8 waves).
// ---------------------------------------------------------------------------
__global__ void __launch_bounds__(256) mha_flash(
    const float* __restrict__ q, const float* __restrict__ kk,
    const float* __restrict__ vv, const float* __restrict__ bias,
    const unsigned char* __restrict__ mask, _Float16* __restrict__ ctx)
{
  __shared__ __align__(16) float    sKf[BN * SKP];     // K chunk [s][d], f32
  __shared__ __align__(16) float    sVf[BN * SVP];     // V chunk [s][d], f32
  __shared__ __align__(16) _Float16 sP[8][16 * SPP];   // per-wave P [m][s]

  const int tid  = threadIdx.x;
  const int wid  = tid >> 5;
  const int lane = tid & 31;
  const int lh   = lane & 15;
  const int hi   = lane >> 4;

  const int blk  = blockIdx.x;
  const int mblk = blk & 7;             // T/BM = 8
  const int h    = (blk >> 3) & 15;
  const int b    = blk >> 7;
  const int m0   = mblk * BM + wid * 16;

  // ---- Q fragments (A layout), f32 -> f16 once, NT (single use) ---------
  const float* qp = q + ((size_t)(b * TSEQ + m0 + lh)) * DM + h * HD;
  v16h qa[2];
  #pragma unroll
  for (int ks = 0; ks < 2; ++ks)
    #pragma unroll
    for (int e = 0; e < 16; ++e)
      qa[ks][e] = (_Float16)__builtin_nontemporal_load(
          &qp[32 * ks + a_kmap(hi, e)]);

  float mrow[8], lrow[8];
  v8f oacc[4];
  #pragma unroll
  for (int r = 0; r < 8; ++r) { mrow[r] = -3.0e38f; lrow[r] = 0.0f; }
  #pragma unroll
  for (int dt = 0; dt < 4; ++dt) { v8f z = {}; oacc[dt] = z; }

  const float* kbase = kk + ((size_t)b * SSEQ) * DM + h * HD;
  const float* vbase = vv + ((size_t)b * SSEQ) * DM + h * HD;
  const float* bbase = bias + (((size_t)b * NH + h) * TSEQ) * SSEQ;
  const unsigned char* mbase = mask + (size_t)b * SSEQ;

  for (int s0 = 0; s0 < SSEQ; s0 += BN) {
    __syncthreads();                    // prior chunk's LDS reads complete

#if HAVE_ASYNC_LDS
    // ---- async DMA stage: K (b128) and V (b64) straight into LDS --------
    #pragma unroll
    for (int it = 0; it < 4; ++it) {    // K: 1024 b128 quads
      int qi = tid + 256 * it;          // 0..1023
      int s = qi >> 4, dq = (qi & 15) * 4;
      __builtin_amdgcn_global_load_async_to_lds_b128(
          (AS1 b128v*)(kbase + (size_t)(s0 + s) * DM + dq),
          (AS3 b128v*)&sKf[s * SKP + dq], 0, 0);
    }
    #pragma unroll
    for (int it = 0; it < 8; ++it) {    // V: 2048 b64 pairs
      int qi = tid + 256 * it;          // 0..2047
      int s = qi >> 5, dq = (qi & 31) * 2;
      __builtin_amdgcn_global_load_async_to_lds_b64(
          (AS1 b64v*)(vbase + (size_t)(s0 + s) * DM + dq),
          (AS3 b64v*)&sVf[s * SVP + dq], 0, 0);
    }
    wait_async0();
#else
    #pragma unroll
    for (int it = 0; it < 16; ++it) {
      int idx = tid + 256 * it;         // 0..4095, coalesced over d
      int s = idx >> 6, d = idx & 63;
      sKf[s * SKP + d] = kbase[(size_t)(s0 + s) * DM + d];
      sVf[s * SVP + d] = vbase[(size_t)(s0 + s) * DM + d];
    }
#endif
    __syncthreads();

    // ---- scores = Q . K^T : 8 WMMAs -------------------------------------
    v8f acc[4];
    #pragma unroll
    for (int nt = 0; nt < 4; ++nt) {
      v8f c = {};
      #pragma unroll
      for (int ks = 0; ks < 2; ++ks) {
        // B fragment: f32 along d from K row (nt*16+lh), d = 8*hi+0..7, 16+8*hi+0..7
        const float* kp = &sKf[(nt * 16 + lh) * SKP + 32 * ks + 8 * hi];
        v16h kb;
        #pragma unroll
        for (int q4 = 0; q4 < 4; ++q4) {
          float4 f = *(const float4*)(kp + 4 * q4 + (q4 >> 1) * 8);
          kb[4 * q4 + 0] = (_Float16)f.x; kb[4 * q4 + 1] = (_Float16)f.y;
          kb[4 * q4 + 2] = (_Float16)f.z; kb[4 * q4 + 3] = (_Float16)f.w;
        }
        c = wmma16(qa[ks], kb, c);
      }
      acc[nt] = c;
    }

    // ---- prefetch next chunk's bias rows (hide behind softmax+PV) ------
    if (s0 + BN < SSEQ) {
      #pragma unroll
      for (int r = 0; r < 8; ++r)
        __builtin_prefetch(
            &bbase[(size_t)(m0 + r + 8 * hi) * SSEQ + s0 + BN + lh], 0, 0);
    }

    // ---- scale, clip, bias (NT: streamed once), mask --------------------
    #pragma unroll
    for (int nt = 0; nt < 4; ++nt) {
      const int sc = s0 + nt * 16 + lh;
      const bool pm = mbase[sc] != 0;
      #pragma unroll
      for (int r = 0; r < 8; ++r) {
        const int m = m0 + r + 8 * hi;
        float bv = __builtin_nontemporal_load(&bbase[(size_t)m * SSEQ + sc]);
        bv = fminf(fmaxf(bv, -10.0f), 10.0f);
        float v0 = acc[nt][r] * 0.125f;
        v0 = fminf(fmaxf(v0, -50.0f), 50.0f) + bv;
        v0 = fminf(fmaxf(v0, -100.0f), 100.0f);
        acc[nt][r] = pm ? -10000.0f : v0;
      }
    }

    // ---- online softmax -------------------------------------------------
    float alpha[8];
    #pragma unroll
    for (int r = 0; r < 8; ++r) {
      float mx = fmaxf(fmaxf(acc[0][r], acc[1][r]),
                       fmaxf(acc[2][r], acc[3][r]));
      #pragma unroll
      for (int off = 1; off < 16; off <<= 1)
        mx = fmaxf(mx, __shfl_xor(mx, off, 16));     // per-row (half-wave)
      const float mnew = fmaxf(mrow[r], mx);
      alpha[r] = __expf(mrow[r] - mnew);
      mrow[r]  = mnew;
    }
    #pragma unroll
    for (int nt = 0; nt < 4; ++nt)
      #pragma unroll
      for (int r = 0; r < 8; ++r)
        acc[nt][r] = __expf(acc[nt][r] - mrow[r]);
    #pragma unroll
    for (int r = 0; r < 8; ++r) {
      float sm = acc[0][r] + acc[1][r] + acc[2][r] + acc[3][r];
      #pragma unroll
      for (int off = 1; off < 16; off <<= 1)
        sm += __shfl_xor(sm, off, 16);
      lrow[r] = lrow[r] * alpha[r] + sm;
      #pragma unroll
      for (int dt = 0; dt < 4; ++dt) oacc[dt][r] *= alpha[r];
    }

    // ---- P: C-layout -> LDS -> A-layout --------------------------------
    #pragma unroll
    for (int nt = 0; nt < 4; ++nt)
      #pragma unroll
      for (int r = 0; r < 8; ++r)
        sP[wid][(r + 8 * hi) * SPP + nt * 16 + lh] = (_Float16)acc[nt][r];
    __syncthreads();   // waves in lockstep; orders sP write->read too

    v16h pa[2];
    #pragma unroll
    for (int ks = 0; ks < 2; ++ks) {
      v8h lo = *(const v8h*)&sP[wid][lh * SPP + 32 * ks + 8 * hi];
      v8h hj = *(const v8h*)&sP[wid][lh * SPP + 32 * ks + 16 + 8 * hi];
      #pragma unroll
      for (int j = 0; j < 8; ++j) { pa[ks][j] = lo[j]; pa[ks][j + 8] = hj[j]; }
    }

    // ---- O += P . V : 8 WMMAs (V gathered f32->f16, conflict-free) ------
    #pragma unroll
    for (int dt = 0; dt < 4; ++dt)
      #pragma unroll
      for (int ss = 0; ss < 2; ++ss) {
        v16h vb;
        #pragma unroll
        for (int j = 0; j < 16; ++j)
          vb[j] = (_Float16)sVf[(32 * ss + 8 * hi + j + (j >> 3) * 8) * SVP + dt * 16 + lh];
        oacc[dt] = wmma16(pa[ss], vb, oacc[dt]);
      }
  }

  // ---- normalize and emit f16 context ------------------------------------
  #pragma unroll
  for (int dt = 0; dt < 4; ++dt)
    #pragma unroll
    for (int r = 0; r < 8; ++r) {
      const int m = m0 + r + 8 * hi;
      const float o = oacc[dt][r] / lrow[r];
      ctx[(size_t)(b * TSEQ + m) * DM + h * HD + dt * 16 + lh] = (_Float16)o;
    }
}

// ---------------------------------------------------------------------------
// Kernel 2: out = ctx @ W^T + b   (2048 x 1024 x 1024), f16 WMMA, f32 out.
// grid = 256 blocks of 256 threads; wave tile 16x64. W/ctx are L2-resident.
// ---------------------------------------------------------------------------
__global__ void __launch_bounds__(256) out_proj(
    const _Float16* __restrict__ ctx, const float* __restrict__ w,
    const float* __restrict__ bvec, float* __restrict__ out)
{
  const int tid  = threadIdx.x;
  const int wid  = tid >> 5;
  const int lane = tid & 31;
  const int lh   = lane & 15;
  const int hi   = lane >> 4;
  const int nb   = blockIdx.x & 15;
  const int mb   = blockIdx.x >> 4;
  const int m0   = mb * 128 + wid * 16;   // rows of [B*T, DM]
  const int n0   = nb * 64;

  v8f acc[4];
  #pragma unroll
  for (int nt = 0; nt < 4; ++nt) {
    const float bj = bvec[n0 + nt * 16 + lh];
    #pragma unroll
    for (int r = 0; r < 8; ++r) acc[nt][r] = bj;    // init with bias
  }

  const _Float16* arow = ctx + (size_t)(m0 + lh) * DM;
  for (int k = 0; k < DM; k += 32) {
    v16h a;
    v8h lo = *(const v8h*)&arow[k + 8 * hi];
    v8h hj = *(const v8h*)&arow[k + 16 + 8 * hi];
    #pragma unroll
    for (int j = 0; j < 8; ++j) { a[j] = lo[j]; a[j + 8] = hj[j]; }

    #pragma unroll
    for (int nt = 0; nt < 4; ++nt) {
      const float* wp = w + (size_t)(n0 + nt * 16 + lh) * DM + k + 8 * hi;
      v16h bf;
      #pragma unroll
      for (int q4 = 0; q4 < 4; ++q4) {
        float4 f = *(const float4*)(wp + 4 * q4 + (q4 >> 1) * 8);
        bf[4 * q4 + 0] = (_Float16)f.x;
        bf[4 * q4 + 1] = (_Float16)f.y;
        bf[4 * q4 + 2] = (_Float16)f.z;
        bf[4 * q4 + 3] = (_Float16)f.w;
      }
      acc[nt] = wmma16(a, bf, acc[nt]);
    }
  }

  #pragma unroll
  for (int nt = 0; nt < 4; ++nt)
    #pragma unroll
    for (int r = 0; r < 8; ++r)
      out[(size_t)(m0 + r + 8 * hi) * DM + n0 + nt * 16 + lh] = acc[nt][r];
}

// ---------------------------------------------------------------------------
extern "C" void kernel_launch(void* const* d_in, const int* in_sizes, int n_in,
                              void* d_out, int out_size, void* d_ws,
                              size_t ws_size, hipStream_t stream) {
  (void)in_sizes; (void)n_in; (void)out_size; (void)ws_size;
  const float*         q     = (const float*)d_in[0];
  const float*         k     = (const float*)d_in[1];
  const float*         v     = (const float*)d_in[2];
  const float*         bias  = (const float*)d_in[3];
  const unsigned char* mask  = (const unsigned char*)d_in[4]; // jax bool = u8
  const float*         out_w = (const float*)d_in[5];
  const float*         out_b = (const float*)d_in[6];
  float*               out   = (float*)d_out;
  _Float16*            ctxws = (_Float16*)d_ws;               // 4 MB f16 ctx

  mha_flash<<<dim3(2 * NH * (TSEQ / BM)), dim3(256), 0, stream>>>(
      q, k, v, bias, mask, ctxws);
  out_proj<<<dim3((2 * TSEQ / 128) * (DM / 64)), dim3(256), 0, stream>>>(
      ctxws, out_w, out_b, out);
}
